// MultiHeadAttention_300647710748
// MI455X (gfx1250) — compile-verified, matrix operand order checked by reading, not run on hardware
//
#include <hip/hip_runtime.h>

// ---------------------------------------------------------------------------
// Problem constants (from reference)
// ---------------------------------------------------------------------------
#define BB   4
#define CC   256
#define NPOS 4096          // HS*WS
#define NH   8
#define NKk  32
#define NVv  32
#define QKV  1088          // NK*NV + NK + NV
#define NT   68            // QKV / 16 column tiles
#define KSTEPS 8           // C / 32

typedef __attribute__((ext_vector_type(16))) __bf16 v16bf;
typedef __attribute__((ext_vector_type(8)))  float  v8f;

union Frag32 { uint4 u[2]; v16bf v; };

__device__ __forceinline__ unsigned short f2bf(float f) {
  unsigned int u = __float_as_uint(f);
  u += 0x7FFFu + ((u >> 16) & 1u);          // round-to-nearest-even
  return (unsigned short)(u >> 16);
}

// ---------------------------------------------------------------------------
// Prep kernel 1: x (B,C,HS,WS) f32 -> xt (B, n, C) bf16  (LDS tiled transpose)
// ---------------------------------------------------------------------------
__global__ void k_transpose_x(const float* __restrict__ x,
                              unsigned short* __restrict__ xt) {
  __shared__ float tile[32][33];
  const int b  = blockIdx.z;
  const int p0 = blockIdx.x * 32;
  const int c0 = blockIdx.y * 32;
  const int tx = threadIdx.x;          // 0..31
  const int ty = threadIdx.y;          // 0..7
  #pragma unroll
  for (int i = 0; i < 32; i += 8)
    tile[ty + i][tx] = x[((size_t)(b * CC + c0 + ty + i) << 12) + p0 + tx];
  __syncthreads();
  #pragma unroll
  for (int i = 0; i < 32; i += 8)
    xt[(((size_t)(b << 12) + p0 + ty + i) << 8) + c0 + tx] = f2bf(tile[tx][ty + i]);
}

// ---------------------------------------------------------------------------
// Prep kernel 2: elementwise f32 -> bf16
// ---------------------------------------------------------------------------
__global__ void k_cvt(const float* __restrict__ in,
                      unsigned short* __restrict__ out, int n) {
  int i = blockIdx.x * 256 + threadIdx.x;
  if (i < n) out[i] = f2bf(in[i]);
}

// ---------------------------------------------------------------------------
// Fused QKV projection + per-position attention.
// Grid: B*NH*(NPOS/64) = 2048 blocks, 256 threads (8 waves).
// Waves: 2 M-groups x 4 N-groups. Each wave: 2 M-tiles x 17 N-tiles.
// Output: a (B*n, NH*NV=256) bf16, row-major.
// ---------------------------------------------------------------------------
__global__ __launch_bounds__(256) void k_attn(
    const unsigned short* __restrict__ xt,
    const unsigned short* __restrict__ wqkv,
    unsigned short* __restrict__ a_out) {
  __shared__ __align__(16) unsigned short xs[64 * 264]; // 64 rows, padded stride
  __shared__ float qs[64 * 32];
  __shared__ float vs[64 * 32];
  __shared__ float qks[64 * 32];

  const int tid   = threadIdx.x;
  const int lane  = tid & 31;
  const int w     = tid >> 5;
  const int wm    = w >> 2;              // 0..1 : M-group (32 rows)
  const int wn    = w & 3;               // 0..3 : N-group
  const int lr    = lane & 15;
  const int hi    = lane >> 4;           // 0/1
  const int kbase = hi * 8;              // per-lane K sub-offset (A/B layout)

  const int blk   = blockIdx.x;
  const int ptile = blk & 63;
  const int h     = (blk >> 6) & 7;
  const int b     = blk >> 9;
  const int pos0  = (b << 12) + (ptile << 6);

  // zero the qk reduction buffer
  for (int i = tid; i < 64 * 32; i += 256) qks[i] = 0.0f;

  // stage x tile (64 x 256 bf16) into LDS with padded row stride 264
  {
    const uint4* xg = (const uint4*)(xt + (size_t)pos0 * CC);
    for (int i = tid; i < 2048; i += 256) {           // 2048 uint4 total
      int row = i >> 5, c = i & 31;                   // 32 uint4 per row
      uint4 val = xg[row * 32 + c];
      *(uint4*)(&xs[row * 264 + c * 8]) = val;
    }
  }
  __syncthreads();

  v8f acc[2][17];
  #pragma unroll
  for (int mt = 0; mt < 2; ++mt)
    #pragma unroll
    for (int i = 0; i < 17; ++i)
      acc[mt][i] = (v8f){0.f,0.f,0.f,0.f,0.f,0.f,0.f,0.f};

  const unsigned short* wbase = wqkv + (size_t)h * QKV * CC;

  #pragma unroll 1
  for (int ks = 0; ks < KSTEPS; ++ks) {
    const int k0 = ks * 32;
    Frag32 af[2];
    #pragma unroll
    for (int mt = 0; mt < 2; ++mt) {
      const unsigned short* p = &xs[(wm * 32 + mt * 16 + lr) * 264 + k0 + kbase];
      af[mt].u[0] = *(const uint4*)p;
      af[mt].u[1] = *(const uint4*)(p + 16);
    }
    #pragma unroll
    for (int i = 0; i < 17; ++i) {
      const int t = wn + 4 * i;
      const unsigned short* wp = wbase + (size_t)(t * 16 + lr) * CC + k0 + kbase;
      Frag32 bfr;
      bfr.u[0] = *(const uint4*)wp;
      bfr.u[1] = *(const uint4*)(wp + 16);
      #pragma unroll
      for (int mt = 0; mt < 2; ++mt)
        acc[mt][i] = __builtin_amdgcn_wmma_f32_16x16x32_bf16(
            false, af[mt].v, false, bfr.v, (short)0, acc[mt][i], false, false);
    }
  }

  // ---- scatter q and v accumulator tiles into LDS ----
  #pragma unroll
  for (int mt = 0; mt < 2; ++mt) {
    #pragma unroll
    for (int i = 0; i < 17; ++i) {
      const int t = wn + 4 * i;
      if (t < 2) {                         // q columns 0..31
        const int kk = t * 16 + lr;
        #pragma unroll
        for (int r = 0; r < 8; ++r) {
          const int row = wm * 32 + mt * 16 + r + hi * 8;
          qs[row * 32 + kk] = acc[mt][i][r];
        }
      } else if (t >= 66) {                // v columns 1056..1087
        const int vv = (t - 66) * 16 + lr;
        #pragma unroll
        for (int r = 0; r < 8; ++r) {
          const int row = wm * 32 + mt * 16 + r + hi * 8;
          vs[row * 32 + vv] = acc[mt][i][r];
        }
      }
    }
  }
  __syncthreads();

  // ---- qk[row][vv] += q[row][kk] * k[row][kk][vv] via LDS float atomics ----
  #pragma unroll
  for (int mt = 0; mt < 2; ++mt) {
    #pragma unroll
    for (int i = 0; i < 17; ++i) {
      const int t = wn + 4 * i;
      if (t >= 2 && t < 66) {
        const int kk = (t - 2) >> 1;
        const int vv = ((t - 2) & 1) * 16 + lr;
        #pragma unroll
        for (int r = 0; r < 8; ++r) {
          const int row = wm * 32 + mt * 16 + r + hi * 8;
          atomicAdd(&qks[row * 32 + vv], qs[row * 32 + kk] * acc[mt][i][r]);
        }
      }
    }
  }
  __syncthreads();

  // ---- softmax over v (32 lanes == 32 v-slots), multiply by v, store bf16 ----
  #pragma unroll 1
  for (int rr = 0; rr < 8; ++rr) {
    const int row = w * 8 + rr;
    float val = qks[row * 32 + lane] * 0.17677669529663687f; // 1/sqrt(NK)
    float m = val;
    #pragma unroll
    for (int off = 16; off >= 1; off >>= 1)
      m = fmaxf(m, __shfl_xor(m, off, 32));
    float e = __expf(val - m);
    float s = e;
    #pragma unroll
    for (int off = 16; off >= 1; off >>= 1)
      s += __shfl_xor(s, off, 32);
    const float res = (e / s) * vs[row * 32 + lane];
    const int gpos = pos0 + row;
    a_out[(size_t)gpos * 256 + h * 32 + lane] = f2bf(res);
  }
}

// ---------------------------------------------------------------------------
// Output projection: out[b,c,pos] = sum_j Wc[c,j] * a[pos,j] + bc[c]
// Computed as D = Wc (M=channels) x a^T (N=positions) so D lands directly in
// channel-major output layout. Grid: 4 channel-tiles x 128 pos-tiles = 512.
// ---------------------------------------------------------------------------
__global__ __launch_bounds__(256) void k_out(
    const unsigned short* __restrict__ a_in,
    const unsigned short* __restrict__ wc,
    const float* __restrict__ bc,
    float* __restrict__ out) {
  const int tid   = threadIdx.x;
  const int lane  = tid & 31;
  const int w     = tid >> 5;
  const int wm    = w >> 2;             // 0..1
  const int wn    = w & 3;              // 0..3
  const int lr    = lane & 15;
  const int hi    = lane >> 4;
  const int kbase = hi * 8;

  const int c0 = (blockIdx.x & 3) * 64;
  const int p0 = (blockIdx.x >> 2) * 128;

  v8f acc[2][2];
  #pragma unroll
  for (int mt = 0; mt < 2; ++mt)
    #pragma unroll
    for (int j = 0; j < 2; ++j)
      acc[mt][j] = (v8f){0.f,0.f,0.f,0.f,0.f,0.f,0.f,0.f};

  #pragma unroll 1
  for (int ks = 0; ks < KSTEPS; ++ks) {
    const int k0 = ks * 32;
    Frag32 af[2], bfr[2];
    #pragma unroll
    for (int mt = 0; mt < 2; ++mt) {
      const unsigned short* p = wc + (size_t)(c0 + wm * 32 + mt * 16 + lr) * CC + k0 + kbase;
      af[mt].u[0] = *(const uint4*)p;
      af[mt].u[1] = *(const uint4*)(p + 16);
    }
    #pragma unroll
    for (int j = 0; j < 2; ++j) {
      const unsigned short* p = a_in + (size_t)(p0 + (wn * 2 + j) * 16 + lr) * 256 + k0 + kbase;
      bfr[j].u[0] = *(const uint4*)p;
      bfr[j].u[1] = *(const uint4*)(p + 16);
    }
    #pragma unroll
    for (int mt = 0; mt < 2; ++mt)
      #pragma unroll
      for (int j = 0; j < 2; ++j)
        acc[mt][j] = __builtin_amdgcn_wmma_f32_16x16x32_bf16(
            false, af[mt].v, false, bfr[j].v, (short)0, acc[mt][j], false, false);
  }

  #pragma unroll
  for (int mt = 0; mt < 2; ++mt) {
    #pragma unroll
    for (int j = 0; j < 2; ++j) {
      #pragma unroll
      for (int r = 0; r < 8; ++r) {
        const int c   = c0 + wm * 32 + mt * 16 + r + hi * 8;
        const int pos = p0 + (wn * 2 + j) * 16 + lr;
        const int bb  = pos >> 12;
        out[((size_t)(bb * CC + c) << 12) + (pos & 4095)] = acc[mt][j][r] + bc[c];
      }
    }
  }
}

// ---------------------------------------------------------------------------
// Launch
// ---------------------------------------------------------------------------
extern "C" void kernel_launch(void* const* d_in, const int* in_sizes, int n_in,
                              void* d_out, int out_size, void* d_ws, size_t ws_size,
                              hipStream_t stream) {
  const float* x    = (const float*)d_in[0];
  const float* wqkv = (const float*)d_in[1];
  const float* wc   = (const float*)d_in[2];
  const float* bc   = (const float*)d_in[3];
  float* out = (float*)d_out;

  char* ws = (char*)d_ws;
  const size_t XT_BYTES   = (size_t)BB * NPOS * CC * 2;        // 8 MiB
  const size_t WQKV_BYTES = (size_t)NH * QKV * CC * 2;         // ~4.25 MiB
  const size_t WC_BYTES   = (size_t)CC * 256 * 2;              // 128 KiB
  unsigned short* xt_bf   = (unsigned short*)(ws);
  unsigned short* wqkv_bf = (unsigned short*)(ws + XT_BYTES);
  unsigned short* wc_bf   = (unsigned short*)(ws + XT_BYTES + WQKV_BYTES);
  unsigned short* a_bf    = (unsigned short*)(ws + XT_BYTES + WQKV_BYTES + WC_BYTES);

  k_transpose_x<<<dim3(NPOS / 32, CC / 32, BB), dim3(32, 8), 0, stream>>>(x, xt_bf);

  const int nwq = NH * QKV * CC;
  k_cvt<<<(nwq + 255) / 256, 256, 0, stream>>>(wqkv, wqkv_bf, nwq);
  const int nwc = CC * 256;
  k_cvt<<<(nwc + 255) / 256, 256, 0, stream>>>(wc, wc_bf, nwc);

  k_attn<<<BB * NH * (NPOS / 64), 256, 0, stream>>>(xt_bf, wqkv_bf, a_bf);

  k_out<<<(CC / 64) * ((BB * NPOS) / 128), 256, 0, stream>>>(a_bf, wc_bf, bc, out);
}
